// X_CrossAttn_60344290509634
// MI455X (gfx1250) — matrix operand order repaired and checked by reading, not compiled, not run on hardware
//
#include <hip/hip_runtime.h>

typedef __attribute__((ext_vector_type(16))) _Float16 v16h;
typedef __attribute__((ext_vector_type(8)))  float    v8f;
typedef _Float16 h16;

#define DEV static __device__ __forceinline__

constexpr int S  = 1024;   // sequence length (both vis and txt)
constexpr int D  = 1024;   // model dim
constexpr int H  = 16;     // heads
constexpr int DK = 64;     // head dim

// ---- WMMA wrapper: D = A(16x32 f16) * B(32x16 f16) + C(f32) ----
DEV v8f wmma16(v16h a, v16h b, v8f c) {
  return __builtin_amdgcn_wmma_f32_16x16x32_f16(
      /*neg_a=*/false, a, /*neg_b=*/false, b,
      /*c_mod=*/(short)0, c, /*reuse_a=*/false, /*reuse_b=*/false);
}

// ---- A-fragment (16x32, f16, row-major source, ld in halves) ----
// lanes 0-15: M=lane, K in {0..7}(elems0-7) and {16..23}(elems8-15)
// lanes16-31: M=lane-16, K in {8..15} and {24..31}
DEV v16h frag_a_h16(const h16* __restrict__ p, int ld) {
  const int lane = threadIdx.x & 31;
  const int r = lane & 15, hi = lane >> 4;
  const h16* row = p + (size_t)r * ld;
  v16h a;
#pragma unroll
  for (int e = 0; e < 8; ++e) a[e] = row[hi * 8 + e];
#pragma unroll
  for (int e = 0; e < 8; ++e) a[8 + e] = row[16 + hi * 8 + e];
  return a;
}

// ---- B-fragment (32x16) from a K-contiguous ("Bt") buffer:
// Bt[n][kk], row stride ld. lane n = lane&15, K in {hi*8..hi*8+7} and {16+hi*8..16+hi*8+7}.
DEV v16h frag_bt_h16(const h16* __restrict__ p, int ld) {
  const int lane = threadIdx.x & 31;
  const int n = lane & 15, hi = lane >> 4;
  const h16* row = p + (size_t)n * ld + hi * 8;
  v16h b;
#pragma unroll
  for (int e = 0; e < 8; ++e) b[e] = row[e];
#pragma unroll
  for (int e = 0; e < 8; ++e) b[8 + e] = row[16 + e];
  return b;
}

// =====================================================================
// Kernel 1: fused QKV/V2 projections (fp32 X, fp32 W -> f16, split heads)
// z=0: Q=vis@Wq  z=1: K=txt@Wk  z=2: V=txt@Wv  z=3: V2=vis@Wv
// grid (B*S/256, D/64, 4), block 256 (8 waves); wave w owns 32 rows
// (2 A-fragments x 4 B-fragments = 8 WMMAs per staged W tile).
// =====================================================================
__global__ __launch_bounds__(256) void k_proj_qkv(
    const float* __restrict__ vis, const float* __restrict__ txt,
    const float* __restrict__ Wq, const float* __restrict__ Wk,
    const float* __restrict__ Wv,
    h16* __restrict__ Qo, h16* __restrict__ Ko,
    h16* __restrict__ Vo, h16* __restrict__ V2o) {
  __shared__ h16 Bt[64][40];  // W tile, transposed: Bt[n][kk]

  const int z = blockIdx.z;
  const float* X = (z == 0 || z == 3) ? vis : txt;
  const float* W = (z == 0) ? Wq : (z == 1) ? Wk : Wv;
  h16* dst = (z == 0) ? Qo : (z == 1) ? Ko : (z == 2) ? Vo : V2o;

  const int row0 = blockIdx.x * 256;
  const int n0   = blockIdx.y * 64;
  const int tid  = threadIdx.x;
  const int w = tid >> 5, lane = tid & 31;
  const int r16 = lane & 15, hi = lane >> 4;

  const float* Arow0 = X + (size_t)(row0 + w * 32 + r16) * D;
  const float* Arow1 = Arow0 + (size_t)16 * D;

  v8f acc0[4] = {};
  v8f acc1[4] = {};
  for (int k0 = 0; k0 < D; k0 += 32) {
    __syncthreads();
#pragma unroll
    for (int i = 0; i < 8; ++i) {          // stage 32x64 W tile (transposed)
      int e = tid + i * 256;
      int c = e >> 6, n = e & 63;          // coalesced over n
      Bt[n][c] = (h16)W[(size_t)(k0 + c) * D + n0 + n];
    }
    __syncthreads();
    // two fp32 -> f16 A-fragments, loaded direct from global
    v16h a0, a1;
#pragma unroll
    for (int e = 0; e < 8; ++e) {
      a0[e]     = (h16)Arow0[k0 + hi * 8 + e];
      a0[8 + e] = (h16)Arow0[k0 + 16 + hi * 8 + e];
      a1[e]     = (h16)Arow1[k0 + hi * 8 + e];
      a1[8 + e] = (h16)Arow1[k0 + 16 + hi * 8 + e];
    }
    // batch all B-fragment loads, then back-to-back WMMAs
    v16h b[4];
#pragma unroll
    for (int j = 0; j < 4; ++j) b[j] = frag_bt_h16(&Bt[j * 16][0], 40);
#pragma unroll
    for (int j = 0; j < 4; ++j) acc0[j] = wmma16(a0, b[j], acc0[j]);
#pragma unroll
    for (int j = 0; j < 4; ++j) acc1[j] = wmma16(a1, b[j], acc1[j]);
  }
  // epilogue: write f16, split-head layout [B][H][S][DK]
#pragma unroll
  for (int u = 0; u < 2; ++u)
#pragma unroll
    for (int j = 0; j < 4; ++j)
#pragma unroll
      for (int e = 0; e < 8; ++e) {
        int rg = row0 + w * 32 + u * 16 + e + 8 * hi;  // row in [0, B*S)
        int cg = n0 + j * 16 + r16;                    // col in [0, INNER)
        int b_ = rg >> 10, s = rg & (S - 1);
        int h  = cg >> 6,  d = cg & (DK - 1);
        float v = u ? acc1[j][e] : acc0[j][e];
        dst[((size_t)(b_ * H + h) * S + s) * DK + d] = (h16)v;
      }
}

// =====================================================================
// Kernel 2: softmax stats m,l per (b,h,q) via online softmax over WMMA
// score tiles. grid (B*H, S/128), block 256; wave owns 16 q rows.
// =====================================================================
__global__ __launch_bounds__(256) void k_stats(
    const h16* __restrict__ Q, const h16* __restrict__ K,
    float* __restrict__ mOut, float* __restrict__ lOut) {
  const int bh = blockIdx.x;
  const int tid = threadIdx.x, w = tid >> 5, lane = tid & 31;
  const int hi = lane >> 4;
  const int q0 = blockIdx.y * 128 + w * 16;

  const h16* Qp = Q + ((size_t)bh * S + q0) * DK;
  const h16* Kp = K + (size_t)bh * S * DK;

  v16h aq0 = frag_a_h16(Qp, DK);        // d 0..31
  v16h aq1 = frag_a_h16(Qp + 32, DK);   // d 32..63
  float m[8], l[8];
#pragma unroll
  for (int e = 0; e < 8; ++e) { m[e] = -1e30f; l[e] = 0.f; }

  for (int kt = 0; kt < S / 16; ++kt) {
    const h16* Kt = Kp + (size_t)kt * 16 * DK;
    v16h b0 = frag_bt_h16(Kt, DK);        // Bt[n][d] == K rows directly
    v16h b1 = frag_bt_h16(Kt + 32, DK);
    v8f sv = {};
    sv = wmma16(aq0, b0, sv);
    sv = wmma16(aq1, b1, sv);
    if (kt + 1 < S / 16)
      __builtin_prefetch(Kt + (size_t)16 * DK, 0, 1);
#pragma unroll
    for (int e = 0; e < 8; ++e) {         // row = e + 8*hi, col = lane&15
      float tm = sv[e];
      tm = fmaxf(tm, __shfl_xor(tm, 1, 32));
      tm = fmaxf(tm, __shfl_xor(tm, 2, 32));
      tm = fmaxf(tm, __shfl_xor(tm, 4, 32));
      tm = fmaxf(tm, __shfl_xor(tm, 8, 32));
      float mn = fmaxf(m[e], tm);
      float ts = __expf(sv[e] - mn);
      ts += __shfl_xor(ts, 1, 32);
      ts += __shfl_xor(ts, 2, 32);
      ts += __shfl_xor(ts, 4, 32);
      ts += __shfl_xor(ts, 8, 32);
      l[e] = l[e] * __expf(m[e] - mn) + ts;
      m[e] = mn;
    }
  }
  if ((lane & 15) == 0) {
#pragma unroll
    for (int e = 0; e < 8; ++e) {
      int q = q0 + e + 8 * hi;
      mOut[(size_t)bh * S + q] = m[e];
      lOut[(size_t)bh * S + q] = l[e];
    }
  }
}

// =====================================================================
// Kernel 3: vis_out core: O1[b,q,h*64+d] = sum_k P[q,k] V[k,d]
// grid (B*H, S/128), block 256. P tiles round-trip through per-wave LDS
// to convert C-layout -> A-fragment layout.
// =====================================================================
__global__ __launch_bounds__(256) void k_attn_vis(
    const h16* __restrict__ Q, const h16* __restrict__ K,
    const h16* __restrict__ V, const float* __restrict__ mSt,
    const float* __restrict__ lSt, h16* __restrict__ O1) {
  __shared__ h16 Vt[64][40];       // V tile transposed: Vt[d][kk]
  __shared__ h16 Ps[8][16][40];    // per-wave P tile 16x32

  const int bh = blockIdx.x;
  const int tid = threadIdx.x, w = tid >> 5, lane = tid & 31;
  const int r16 = lane & 15, hi = lane >> 4;
  const int q0 = blockIdx.y * 128 + w * 16;

  const h16* Qp = Q + ((size_t)bh * S + q0) * DK;
  const h16* Kp = K + (size_t)bh * S * DK;
  const h16* Vp = V + (size_t)bh * S * DK;

  v16h aq0 = frag_a_h16(Qp, DK);
  v16h aq1 = frag_a_h16(Qp + 32, DK);
  float mreg[8], linv[8];
#pragma unroll
  for (int e = 0; e < 8; ++e) {
    int q = q0 + e + 8 * hi;
    mreg[e] = mSt[(size_t)bh * S + q];
    linv[e] = 1.0f / lSt[(size_t)bh * S + q];
  }

  v8f acc[4] = {};
  for (int k0 = 0; k0 < S; k0 += 32) {
    __syncthreads();
#pragma unroll
    for (int i = 0; i < 8; ++i) {        // stage V tile 32x64, transposed
      int e = tid + i * 256;
      int c = e >> 6, d = e & 63;
      Vt[d][c] = Vp[(size_t)(k0 + c) * DK + d];
    }
#pragma unroll
    for (int t = 0; t < 2; ++t) {        // two 16-col score subtiles
      const h16* Kt = Kp + (size_t)(k0 + t * 16) * DK;
      v16h b0 = frag_bt_h16(Kt, DK);
      v16h b1 = frag_bt_h16(Kt + 32, DK);
      v8f sv = {};
      sv = wmma16(aq0, b0, sv);
      sv = wmma16(aq1, b1, sv);
#pragma unroll
      for (int e = 0; e < 8; ++e) {
        float p = __expf(sv[e] - mreg[e]) * linv[e];
        Ps[w][e + 8 * hi][t * 16 + r16] = (h16)p;
      }
    }
    __syncthreads();
    v16h a = frag_a_h16(&Ps[w][0][0], 40);   // P as A-fragment
    v16h b[4];
#pragma unroll
    for (int j = 0; j < 4; ++j) b[j] = frag_bt_h16(&Vt[j * 16][0], 40);
#pragma unroll
    for (int j = 0; j < 4; ++j) acc[j] = wmma16(a, b[j], acc[j]);
    if (k0 + 32 < S)
      __builtin_prefetch(Kp + (size_t)(k0 + 32) * DK, 0, 1);
  }
  const int b = bh >> 4, h = bh & (H - 1);
#pragma unroll
  for (int j = 0; j < 4; ++j)
#pragma unroll
    for (int e = 0; e < 8; ++e) {
      int q = q0 + e + 8 * hi;
      int d = j * 16 + r16;
      O1[((size_t)(b * S + q)) * D + h * DK + d] = (h16)acc[j][e];
    }
}

// =====================================================================
// Kernel 4: txt_out core: O2[b,k,h*64+d] = sum_q P[q,k] V2[q,d]
// Computes S^T tiles directly (A = K rows, B = Q rows), so stats are
// per-column => one scalar per lane. grid (B*H, S/128), block 256.
// =====================================================================
__global__ __launch_bounds__(256) void k_attn_txt(
    const h16* __restrict__ Q, const h16* __restrict__ K,
    const h16* __restrict__ V2, const float* __restrict__ mSt,
    const float* __restrict__ lSt, h16* __restrict__ O2) {
  __shared__ h16 V2t[64][40];      // V2 tile transposed: V2t[d][qq]
  __shared__ h16 Ps[8][16][40];    // per-wave P^T tile 16x32

  const int bh = blockIdx.x;
  const int tid = threadIdx.x, w = tid >> 5, lane = tid & 31;
  const int r16 = lane & 15, hi = lane >> 4;
  const int k0w = blockIdx.y * 128 + w * 16;

  const h16* Qp  = Q  + (size_t)bh * S * DK;
  const h16* Kp  = K  + ((size_t)bh * S + k0w) * DK;
  const h16* V2p = V2 + (size_t)bh * S * DK;

  v16h ak0 = frag_a_h16(Kp, DK);
  v16h ak1 = frag_a_h16(Kp + 32, DK);

  v8f acc[4] = {};
  for (int q0 = 0; q0 < S; q0 += 32) {
    __syncthreads();
#pragma unroll
    for (int i = 0; i < 8; ++i) {        // stage V2 tile 32x64, transposed
      int e = tid + i * 256;
      int c = e >> 6, d = e & 63;
      V2t[d][c] = V2p[(size_t)(q0 + c) * DK + d];
    }
#pragma unroll
    for (int t = 0; t < 2; ++t) {
      const int qs = q0 + t * 16;
      v16h b0 = frag_bt_h16(Qp + (size_t)qs * DK, DK);
      v16h b1 = frag_bt_h16(Qp + (size_t)qs * DK + 32, DK);
      v8f sv = {};                        // S^T tile: rows=k, cols=q
      sv = wmma16(ak0, b0, sv);
      sv = wmma16(ak1, b1, sv);
      float mq = mSt[(size_t)bh * S + qs + r16];   // per-column stats
      float li = 1.0f / lSt[(size_t)bh * S + qs + r16];
#pragma unroll
      for (int e = 0; e < 8; ++e) {
        float p = __expf(sv[e] - mq) * li;
        Ps[w][e + 8 * hi][t * 16 + r16] = (h16)p;
      }
    }
    __syncthreads();
    v16h a = frag_a_h16(&Ps[w][0][0], 40);   // P^T as A-fragment
    v16h b[4];
#pragma unroll
    for (int j = 0; j < 4; ++j) b[j] = frag_bt_h16(&V2t[j * 16][0], 40);
#pragma unroll
    for (int j = 0; j < 4; ++j) acc[j] = wmma16(a, b[j], acc[j]);
    if (q0 + 32 < S)
      __builtin_prefetch(Qp + (size_t)(q0 + 32) * DK, 0, 1);
  }
  const int b = bh >> 4, h = bh & (H - 1);
#pragma unroll
  for (int j = 0; j < 4; ++j)
#pragma unroll
    for (int e = 0; e < 8; ++e) {
      int k = k0w + e + 8 * hi;
      int d = j * 16 + r16;
      O2[((size_t)(b * S + k)) * D + h * DK + d] = (h16)acc[j][e];
    }
}

// =====================================================================
// Kernel 5: output projection: out = Om @ Wo (fp32 out)
// z=0: vis_out from O1, z=1: txt_out from O2.
// grid (B*S/256, D/64, 2), block 256; wave owns 32 rows (8 WMMAs/tile).
// =====================================================================
__global__ __launch_bounds__(256) void k_proj_out(
    const h16* __restrict__ O1, const h16* __restrict__ O2,
    const float* __restrict__ Wo,
    float* __restrict__ outVis, float* __restrict__ outTxt) {
  __shared__ h16 Bt[64][40];

  const h16* A = (blockIdx.z == 0) ? O1 : O2;
  float* dst = (blockIdx.z == 0) ? outVis : outTxt;

  const int row0 = blockIdx.x * 256;
  const int n0   = blockIdx.y * 64;
  const int tid  = threadIdx.x;
  const int w = tid >> 5, lane = tid & 31;
  const int r16 = lane & 15, hi = lane >> 4;

  const h16* Arow0 = A + (size_t)(row0 + w * 32 + r16) * D;
  const h16* Arow1 = Arow0 + (size_t)16 * D;

  v8f acc0[4] = {};
  v8f acc1[4] = {};
  for (int k0 = 0; k0 < D; k0 += 32) {
    __syncthreads();
#pragma unroll
    for (int i = 0; i < 8; ++i) {
      int e = tid + i * 256;
      int c = e >> 6, n = e & 63;
      Bt[n][c] = (h16)Wo[(size_t)(k0 + c) * D + n0 + n];
    }
    __syncthreads();
    v16h a0 = frag_a_h16(Arow0 + k0, D);
    v16h a1 = frag_a_h16(Arow1 + k0, D);
    v16h b[4];
#pragma unroll
    for (int j = 0; j < 4; ++j) b[j] = frag_bt_h16(&Bt[j * 16][0], 40);
#pragma unroll
    for (int j = 0; j < 4; ++j) acc0[j] = wmma16(a0, b[j], acc0[j]);
#pragma unroll
    for (int j = 0; j < 4; ++j) acc1[j] = wmma16(a1, b[j], acc1[j]);
  }
#pragma unroll
  for (int u = 0; u < 2; ++u)
#pragma unroll
    for (int j = 0; j < 4; ++j)
#pragma unroll
      for (int e = 0; e < 8; ++e) {
        int rg = row0 + w * 32 + u * 16 + e + 8 * hi;
        int cg = n0 + j * 16 + r16;
        dst[(size_t)rg * D + cg] = u ? acc1[j][e] : acc0[j][e];
      }
}

// =====================================================================
extern "C" void kernel_launch(void* const* d_in, const int* in_sizes, int n_in,
                              void* d_out, int out_size, void* d_ws, size_t ws_size,
                              hipStream_t stream) {
  const float* vis = (const float*)d_in[0];
  const float* txt = (const float*)d_in[1];
  const float* Wq  = (const float*)d_in[2];
  const float* Wk  = (const float*)d_in[3];
  const float* Wv  = (const float*)d_in[4];
  const float* Wo  = (const float*)d_in[5];

  const int B = in_sizes[0] / (S * D);          // batch (4)
  const size_t tokens = (size_t)B * S;          // 4096

  // workspace layout (f16 tensors + f32 stats)
  char* ws = (char*)d_ws;
  const size_t tsz = tokens * D * sizeof(h16);  // 8 MB per [B,S,1024] f16
  h16* Qw  = (h16*)(ws + 0 * tsz);              // [B,H,S,DK]
  h16* Kw  = (h16*)(ws + 1 * tsz);
  h16* Vw  = (h16*)(ws + 2 * tsz);
  h16* V2w = (h16*)(ws + 3 * tsz);
  h16* O1  = (h16*)(ws + 4 * tsz);              // merged [B,S,INNER]
  h16* O2  = (h16*)(ws + 5 * tsz);
  float* mSt = (float*)(ws + 6 * tsz);
  float* lSt = mSt + tokens * H;

  float* outVis = (float*)d_out;
  float* outTxt = outVis + tokens * D;

  dim3 blk(256);
  // 1) projections
  k_proj_qkv<<<dim3(tokens / 256, D / 64, 4), blk, 0, stream>>>(
      vis, txt, Wq, Wk, Wv, Qw, Kw, Vw, V2w);
  // 2) softmax stats
  k_stats<<<dim3(B * H, S / 128), blk, 0, stream>>>(Qw, Kw, mSt, lSt);
  // 3) vis attention
  k_attn_vis<<<dim3(B * H, S / 128), blk, 0, stream>>>(Qw, Kw, Vw, mSt, lSt, O1);
  // 4) txt attention (transposed P, not renormalized)
  k_attn_txt<<<dim3(B * H, S / 128), blk, 0, stream>>>(Qw, Kw, V2w, mSt, lSt, O2);
  // 5) output projections
  k_proj_out<<<dim3(tokens / 256, D / 64, 2), blk, 0, stream>>>(
      O1, O2, Wo, outVis, outTxt);
}
